// GraphHeadAttention_42202348650871
// MI455X (gfx1250) — matrix operand order repaired and checked by reading, not compiled, not run on hardware
//
#include <hip/hip_runtime.h>

#define BB   8
#define NN   1024
#define DIMM 768
#define HH   12
#define DHH  64
#define TOPK 16
#define QKV3 2304   // 3*H*DH

typedef __attribute__((ext_vector_type(16))) _Float16 v16h;
typedef __attribute__((ext_vector_type(8)))  _Float16 v8h;
typedef __attribute__((ext_vector_type(8)))  float    v8f;

__device__ __forceinline__ v8f wmma32(v16h a, v16h b, v8f c) {
  // v_wmma_f32_16x16x32_f16: D = A(16x32,f16) x B(32x16,f16) + C(16x16,f32)
  return __builtin_amdgcn_wmma_f32_16x16x32_f16(false, a, false, b, (short)0, c,
                                                false, false);
}

// A fragment (16x32 f16, row-major, leading dim lda), tile at (row0, k0).
// Lane L (half h=L>>4): row=row0+(L&15); elements e -> K=k0+16h+e
// => two contiguous 16B vector loads.
__device__ __forceinline__ v16h load_a_frag(const _Float16* A, int lda,
                                            int row0, int k0, int lane) {
  int r  = row0 + (lane & 15);
  int hh = (lane >> 4) & 1;
  const _Float16* p = A + (size_t)r * lda + k0 + 16 * hh;
  v8h lo = *(const v8h*)(p);
  v8h hi = *(const v8h*)(p + 8);
  v16h a;
#pragma unroll
  for (int e = 0; e < 8; ++e) { a[e] = lo[e]; a[e + 8] = hi[e]; }
  return a;
}

// B fragment from a K-contiguous (column-major) matrix BT[ncols][ldk]:
// lane's column = col0+(lane&15); 16 contiguous K elements at k0+16h.
// One 32-byte contiguous load per lane.
__device__ __forceinline__ v16h load_bT_frag(const _Float16* BT, int ldk,
                                             int k0, int col0, int lane) {
  int c  = col0 + (lane & 15);
  int hh = (lane >> 4) & 1;
  return *(const v16h*)(BT + (size_t)c * ldk + k0 + 16 * hh);
}

// ---------------- LayerNorm (fp32 in, f16 out) ----------------
__global__ void ln_kernel(const float* __restrict__ x, const float* __restrict__ g,
                          const float* __restrict__ bta, _Float16* __restrict__ xh) {
  int row = blockIdx.x;
  const float* xr = x + (size_t)row * DIMM;
  float s = 0.f, s2 = 0.f;
  for (int i = threadIdx.x; i < DIMM; i += 256) { float v = xr[i]; s += v; s2 += v * v; }
  __shared__ float sh1[256], sh2[256];
  sh1[threadIdx.x] = s; sh2[threadIdx.x] = s2;
  __syncthreads();
  for (int o = 128; o > 0; o >>= 1) {
    if (threadIdx.x < o) { sh1[threadIdx.x] += sh1[threadIdx.x + o];
                           sh2[threadIdx.x] += sh2[threadIdx.x + o]; }
    __syncthreads();
  }
  float mu  = sh1[0] * (1.f / DIMM);
  float var = sh2[0] * (1.f / DIMM) - mu * mu;
  float inv = rsqrtf(var + 1e-5f);
  for (int i = threadIdx.x; i < DIMM; i += 256)
    xh[(size_t)row * DIMM + i] = (_Float16)(((xr[i] - mu) * inv) * g[i] + bta[i]);
}

// ---------------- fp32 [K,N] -> f16 transposed [N,K] (LDS-tiled) ----------------
__global__ void transpose_f32_to_f16(const float* __restrict__ src,
                                     _Float16* __restrict__ dst, int K, int N) {
  __shared__ float tile[32][33];
  int k0 = blockIdx.y * 32, c0 = blockIdx.x * 32;
  int tx = threadIdx.x & 31, ty = threadIdx.x >> 5;   // 256 threads = 32x8
  for (int i = ty; i < 32; i += 8)
    tile[i][tx] = src[(size_t)(k0 + i) * N + (c0 + tx)];
  __syncthreads();
  for (int i = ty; i < 32; i += 8)
    dst[(size_t)(c0 + i) * K + (k0 + tx)] = (_Float16)tile[tx][i];
}

// ---------------- QKV projection: [8192,768] x [768,2304] ----------------
// B operand from K-contiguous wT[2304][768]. Each wave: 16x32 output tile.
// Scatters Q (pre-scaled), K row-major [B,H,N,DH]; V transposed [B,H,DH,N].
__global__ void qkv_gemm_kernel(const _Float16* __restrict__ xh,
                                const _Float16* __restrict__ wT,
                                _Float16* __restrict__ qh, _Float16* __restrict__ kh,
                                _Float16* __restrict__ vt) {
  const int tilesN = QKV3 / 32;                  // 72 (32-wide tiles)
  int lane = threadIdx.x & 31, wv = threadIdx.x >> 5;
  int tile = blockIdx.x * 8 + wv;
  int tm = tile / tilesN, tn = tile % tilesN;
  v8f acc0 = {}, acc1 = {};
  for (int k0 = 0; k0 < DIMM; k0 += 32) {
    v16h a  = load_a_frag(xh, DIMM, tm * 16, k0, lane);
    v16h b0 = load_bT_frag(wT, DIMM, k0, tn * 32, lane);
    v16h b1 = load_bT_frag(wT, DIMM, k0, tn * 32 + 16, lane);
    acc0 = wmma32(a, b0, acc0);
    acc1 = wmma32(a, b1, acc1);
  }
  int hh = (lane >> 4) & 1;
#pragma unroll
  for (int half = 0; half < 2; ++half) {
    int gc = tn * 32 + half * 16 + (lane & 15);
    int sec = gc / DIMM;                         // 0=q 1=k 2=v
    int cc  = gc - sec * DIMM;
    int head = cc >> 6, d = cc & 63;
    const v8f& acc = half ? acc1 : acc0;
#pragma unroll
    for (int r = 0; r < 8; ++r) {
      int gr = tm * 16 + r + 8 * hh;
      int bb = gr >> 10, n = gr & 1023;
      float v = acc[r];
      if (sec == 0) {
        qh[(((size_t)bb * HH + head) * NN + n) * DHH + d] = (_Float16)(v * 0.125f);
      } else if (sec == 1) {
        kh[(((size_t)bb * HH + head) * NN + n) * DHH + d] = (_Float16)v;
      } else {
        vt[(((size_t)bb * HH + head) * DHH + d) * NN + n] = (_Float16)v;  // transposed
      }
    }
  }
}

// ---------------- Fused attention: one block per (b*H, 16-query tile) ----------------
__global__ void attn_kernel(const _Float16* __restrict__ qh, const _Float16* __restrict__ kh,
                            const _Float16* __restrict__ vt, const float* __restrict__ alpha_p,
                            _Float16* __restrict__ ao) {
  extern __shared__ char smem[];
  float*    dots = (float*)smem;                          // 16*1024 f32 = 64KB
  _Float16* aw   = (_Float16*)(smem + 16 * NN * 4);       // 16*1024 f16 = 32KB
  int lane = threadIdx.x & 31, wv = threadIdx.x >> 5;
  int hh = (lane >> 4) & 1;
  int bh = blockIdx.x >> 6;        // b*H + h
  int qt = blockIdx.x & 63;        // query tile of 16 rows
  const _Float16* Q  = qh + (size_t)bh * NN * DHH;
  const _Float16* Kt = kh + (size_t)bh * NN * DHH;   // [N][DH]: K-contiguous for QK^T
  const _Float16* Vt = vt + (size_t)bh * DHH * NN;   // [DH][N]: K-contiguous for attn@V

  // Phase 1: dots[16,1024] = Q_tile(16x64) * K^T (two K=32 WMMA steps per tile).
  v16h a0 = load_a_frag(Q, DHH, qt * 16, 0, lane);
  v16h a1 = load_a_frag(Q, DHH, qt * 16, 32, lane);
  for (int nt = wv; nt < NN / 16; nt += 8) {
    int col = nt * 16 + (lane & 15);            // key index
    const _Float16* kp = Kt + (size_t)col * DHH;
    v16h b0 = *(const v16h*)(kp + 16 * hh);       // K rows 0..31
    v16h b1 = *(const v16h*)(kp + 32 + 16 * hh);  // K rows 32..63
    v8f acc = {};
    acc = wmma32(a0, b0, acc);
    acc = wmma32(a1, b1, acc);
#pragma unroll
    for (int r = 0; r < 8; ++r)
      dots[(size_t)(r + 8 * hh) * NN + col] = acc[r];
  }
  __syncthreads();

  // Phase 2: per-row stats + blended weights. 8 waves x 2 rows.
  float al = alpha_p[0];
  al = fminf(fmaxf(al, 0.f), 1.f);
  for (int rr = 0; rr < 2; ++rr) {
    int m = wv + 8 * rr;
    float* drow = dots + (size_t)m * NN;
    float mx = -3.0e38f;
    for (int i = lane; i < NN; i += 32) mx = fmaxf(mx, drow[i]);
#pragma unroll
    for (int o = 16; o; o >>= 1) mx = fmaxf(mx, __shfl_xor(mx, o, 32));
    float se = 0.f;
    for (int i = lane; i < NN; i += 32) se += __expf(drow[i] - mx);
#pragma unroll
    for (int o = 16; o; o >>= 1) se += __shfl_xor(se, o, 32);
    // iterative top-16: descend through the 16 largest values
    float bound = 3.0e38f, st = 0.f;
    for (int t = 0; t < TOPK; ++t) {
      float lm = -3.0e38f;
      for (int i = lane; i < NN; i += 32) {
        float v = drow[i];
        lm = (v < bound) ? fmaxf(lm, v) : lm;
      }
#pragma unroll
      for (int o = 16; o; o >>= 1) lm = fmaxf(lm, __shfl_xor(lm, o, 32));
      st += __expf(lm - mx);
      bound = lm;
    }
    float thr = bound;                           // 16th-largest value
    float gi = al / se, si = (1.f - al) / st;
    for (int i = lane; i < NN; i += 32) {
      float e = __expf(drow[i] - mx);
      float w = gi * e;
      if (drow[i] >= thr) w += si * e;           // sparse softmax contribution
      aw[(size_t)m * NN + i] = (_Float16)w;
    }
  }
  __syncthreads();

  // Phase 3: out(16x64) = aw(16x1024) * V(1024x64). Waves 0..3 own 16-col tiles.
  if (wv < 4) {
    int col = wv * 16 + (lane & 15);             // d index within head
    v8f acc = {};
    for (int k0 = 0; k0 < NN; k0 += 32) {
      v16h a = load_a_frag(aw, NN, 0, k0, lane);
      v16h b = *(const v16h*)(Vt + (size_t)col * NN + k0 + 16 * hh);  // contiguous
      acc = wmma32(a, b, acc);
    }
    int bb = bh / HH, hd = bh % HH;
#pragma unroll
    for (int r = 0; r < 8; ++r) {
      int n = qt * 16 + r + 8 * hh;
      ao[((size_t)bb * NN + n) * (HH * DHH) + hd * DHH + col] = (_Float16)acc[r];
    }
  }
}

// ---------------- Output projection: [8192,768] x [768,768] + bias ----------------
__global__ void out_gemm_kernel(const _Float16* __restrict__ ah,
                                const _Float16* __restrict__ wT,
                                const float* __restrict__ bias,
                                float* __restrict__ out) {
  const int tilesN = DIMM / 32;                  // 24 (32-wide tiles)
  int lane = threadIdx.x & 31, wv = threadIdx.x >> 5;
  int tile = blockIdx.x * 8 + wv;
  int tm = tile / tilesN, tn = tile % tilesN;
  v8f acc0 = {}, acc1 = {};
  for (int k0 = 0; k0 < DIMM; k0 += 32) {
    v16h a  = load_a_frag(ah, DIMM, tm * 16, k0, lane);
    v16h b0 = load_bT_frag(wT, DIMM, k0, tn * 32, lane);
    v16h b1 = load_bT_frag(wT, DIMM, k0, tn * 32 + 16, lane);
    acc0 = wmma32(a, b0, acc0);
    acc1 = wmma32(a, b1, acc1);
  }
  int hh = (lane >> 4) & 1;
#pragma unroll
  for (int half = 0; half < 2; ++half) {
    int gc = tn * 32 + half * 16 + (lane & 15);
    float bv = bias[gc];
    const v8f& acc = half ? acc1 : acc0;
#pragma unroll
    for (int r = 0; r < 8; ++r) {
      int gr = tm * 16 + r + 8 * hh;
      out[(size_t)gr * DIMM + gc] = acc[r] + bv;
    }
  }
}

extern "C" void kernel_launch(void* const* d_in, const int* in_sizes, int n_in,
                              void* d_out, int out_size, void* d_ws, size_t ws_size,
                              hipStream_t stream) {
  const float* x     = (const float*)d_in[0];
  const float* lng   = (const float*)d_in[1];
  const float* lnb   = (const float*)d_in[2];
  const float* wqkv  = (const float*)d_in[3];
  const float* wout  = (const float*)d_in[4];
  const float* bout  = (const float*)d_in[5];
  const float* alpha = (const float*)d_in[6];
  float* out = (float*)d_out;

  char* ws = (char*)d_ws;
  size_t off = 0;
  auto alloc = [&](size_t bytes) -> void* {
    void* p = ws + off;
    off += (bytes + 255) & ~(size_t)255;
    return p;
  };
  _Float16* xh  = (_Float16*)alloc((size_t)BB * NN * DIMM * 2);       // LN output f16
  _Float16* wqT = (_Float16*)alloc((size_t)DIMM * QKV3 * 2);          // w_qkv^T f16 [2304][768]
  _Float16* woT = (_Float16*)alloc((size_t)DIMM * DIMM * 2);          // w_out^T f16 [768][768]
  _Float16* qhp = (_Float16*)alloc((size_t)BB * HH * NN * DHH * 2);   // Q (scaled) [B,H,N,DH]
  _Float16* khp = (_Float16*)alloc((size_t)BB * HH * NN * DHH * 2);   // K [B,H,N,DH]
  _Float16* vtp = (_Float16*)alloc((size_t)BB * HH * NN * DHH * 2);   // V^T [B,H,DH,N]
  _Float16* aoh = (_Float16*)alloc((size_t)BB * NN * DIMM * 2);       // attn output f16

  hipLaunchKernelGGL(ln_kernel, dim3(BB * NN), dim3(256), 0, stream, x, lng, lnb, xh);

  hipLaunchKernelGGL(transpose_f32_to_f16, dim3(QKV3 / 32, DIMM / 32), dim3(256), 0,
                     stream, wqkv, wqT, DIMM, QKV3);
  hipLaunchKernelGGL(transpose_f32_to_f16, dim3(DIMM / 32, DIMM / 32), dim3(256), 0,
                     stream, wout, woT, DIMM, DIMM);

  hipLaunchKernelGGL(qkv_gemm_kernel,
                     dim3((BB * NN / 16) * (QKV3 / 32) / 8), dim3(256), 0, stream,
                     xh, wqT, qhp, khp, vtp);

  size_t smem = (size_t)16 * NN * 4 + (size_t)16 * NN * 2;   // 96 KB dynamic LDS
  hipLaunchKernelGGL(attn_kernel, dim3(BB * HH * (NN / 16)), dim3(256), smem, stream,
                     qhp, khp, vtp, alpha, aoh);

  hipLaunchKernelGGL(out_gemm_kernel,
                     dim3((BB * NN / 16) * (DIMM / 32) / 8), dim3(256), 0, stream,
                     aoh, woT, bout, out);
}
